// MoE_16441134809274
// MI455X (gfx1250) — matrix operand order repaired and checked by reading, not compiled, not run on hardware
//
#include <hip/hip_runtime.h>

// MoE: G=8 S=2048 D=256 E=8 H=512 K=2  (fixed per reference setup_inputs)
#define GQ 8
#define SQ 2048
#define DQ 256
#define EQ 8
#define HQ 512
#define BQ (GQ * SQ)            // 16384 tokens
#define CAPQ (BQ * 2)           // worst-case routed entries per expert
#define MAX_TILES (CAPQ / 16)   // 2048 16-token tiles per expert
#define RMS_EPS 1e-6f

typedef __attribute__((ext_vector_type(16))) __bf16 v16bf;
typedef __attribute__((ext_vector_type(8)))  float  v8f;

// ---------- helpers ----------

__device__ __forceinline__ unsigned short f2bf(float f) {
    union { float f; unsigned int u; } v;
    v.f = f;
    unsigned int u = v.u;
    // round-to-nearest-even
    u += 0x7FFFu + ((u >> 16) & 1u);
    return (unsigned short)(u >> 16);
}

__device__ __forceinline__ float gelu_tanh(float x) {
    // jax.nn.gelu default (tanh approximation)
    float x3 = x * x * x;
    return 0.5f * x * (1.0f + tanhf(0.7978845608028654f * (x + 0.044715f * x3)));
}

// 16-bit A-matrix 16x32 lane layout (ISA 7.12.2):
// lane<16 holds K={0..7,16..23}, lane>=16 holds K={8..15,24..31}; r=0..15 slot.
__device__ __forceinline__ int a_koff(int L, int r) {
    return (r & 7) + ((L >> 4) << 3) + ((r >> 3) << 4);
}

// ---------- kernel 0: zero output + expert counters ----------

__global__ void zero_kernel(float* out, int n, int* counts) {
    int i = blockIdx.x * blockDim.x + threadIdx.x;
    int stride = gridDim.x * blockDim.x;
    for (; i < n; i += stride) out[i] = 0.0f;
    if (blockIdx.x == 0 && threadIdx.x < EQ) counts[threadIdx.x] = 0;
}

// ---------- kernel 1: pack gating weights (E,2,H,D) fp32 -> bf16 B-operand tiles ----------
// pg[e][j][ft(H/16)][kc(D/32)][lane(32)][r(16)] :
//   col n = ft*16 + lane%16 (f), K = kc*32 + 8*(lane/16) + (r%8) + 16*(r/8) (d)

__global__ void pack_gating(const float* __restrict__ wg, unsigned short* __restrict__ pg) {
    const int n = EQ * 2 * (HQ / 16) * (DQ / 32) * 32 * 16;
    int i = blockIdx.x * blockDim.x + threadIdx.x;
    int stride = gridDim.x * blockDim.x;
    for (; i < n; i += stride) {
        int t = i;
        int r  = t & 15;  t >>= 4;
        int L  = t & 31;  t >>= 5;
        int kc = t & 7;   t >>= 3;
        int ft = t & 31;  t >>= 5;
        int j  = t & 1;   int e = t >> 1;
        int f = ft * 16 + (L & 15);
        int d = kc * 32 + ((L >> 4) << 3) + (r & 7) + ((r >> 3) << 4);
        pg[i] = f2bf(wg[(((e * 2 + j) * HQ + f) * DQ) + d]);
    }
}

// ---------- kernel 2: pack linear weights (E,H,D) fp32 -> bf16 B-operand tiles ----------
// pl[e][dt(D/16)][kc2(H/32)][lane(32)][r(16)] :
//   col n = dt*16 + lane%16 (d), K = kc2*32 + 8*(lane/16) + (r%8) + 16*(r/8) (f)

__global__ void pack_linear(const float* __restrict__ wl, unsigned short* __restrict__ pl) {
    const int n = EQ * (DQ / 16) * (HQ / 32) * 32 * 16;
    int i = blockIdx.x * blockDim.x + threadIdx.x;
    int stride = gridDim.x * blockDim.x;
    for (; i < n; i += stride) {
        int t = i;
        int r   = t & 15; t >>= 4;
        int L   = t & 31; t >>= 5;
        int kc2 = t & 15; t >>= 4;
        int dt  = t & 15; int e = t >> 4;
        int f = kc2 * 32 + ((L >> 4) << 3) + (r & 7) + ((r >> 3) << 4);
        int d = dt * 16 + (L & 15);
        pl[i] = f2bf(wl[(e * HQ + f) * DQ + d]);
    }
}

// ---------- kernel 3: router (RMSnorm -> logits -> softmax -> top2 -> bucket append) ----------

__global__ __launch_bounds__(256) void router_kernel(
    const float* __restrict__ x, const float* __restrict__ wr,
    const float* __restrict__ rscale,
    int* counts, int* btok, float* bwt)
{
    __shared__ float red[DQ];
    __shared__ float lg[EQ];
    int tid = threadIdx.x;
    int token = blockIdx.x;

    float v = x[(long)token * DQ + tid];
    red[tid] = v * v;
    __syncthreads();
    for (int s = 128; s > 0; s >>= 1) {
        if (tid < s) red[tid] += red[tid + s];
        __syncthreads();
    }
    float var = red[0] * (1.0f / DQ);
    float ri = v * rsqrtf(var + RMS_EPS) * rsqrtf((float)DQ) * rscale[tid];

    if (tid < EQ) lg[tid] = 0.0f;
    __syncthreads();
#pragma unroll
    for (int e = 0; e < EQ; ++e)
        atomicAdd(&lg[e], ri * wr[tid * EQ + e]);
    __syncthreads();

    if (tid == 0) {
        float p[EQ];
        float mx = lg[0];
#pragma unroll
        for (int e = 1; e < EQ; ++e) mx = fmaxf(mx, lg[e]);
        float sum = 0.0f;
#pragma unroll
        for (int e = 0; e < EQ; ++e) { p[e] = __expf(lg[e] - mx); sum += p[e]; }
        float inv = 1.0f / sum;
#pragma unroll
        for (int e = 0; e < EQ; ++e) p[e] *= inv;

        int e1 = 0; float p1 = p[0];
#pragma unroll
        for (int e = 1; e < EQ; ++e) if (p[e] > p1) { p1 = p[e]; e1 = e; }
        int e2 = -1; float p2 = -1.0f;
#pragma unroll
        for (int e = 0; e < EQ; ++e) if (e != e1 && p[e] > p2) { p2 = p[e]; e2 = e; }

        float renorm = p1 + p2;
        float w1 = p1 / renorm, w2 = p2 / renorm;

        int pos1 = atomicAdd(&counts[e1], 1);
        btok[e1 * CAPQ + pos1] = token;
        bwt [e1 * CAPQ + pos1] = w1;
        int pos2 = atomicAdd(&counts[e2], 1);
        btok[e2 * CAPQ + pos2] = token;
        bwt [e2 * CAPQ + pos2] = w2;
    }
}

// ---------- kernel 4: expert FFN via WMMA bf16 ----------
// One block = (expert, 16-token tile). 256 threads = 8 wave32.
// GEMM1: [16 x 256] x [256 x 1024]  (x1|x2, K=D)  -> gelu(x1)*x2
// GEMM2: [16 x 512] x [512 x 256]   (K=H)         -> scaled atomic combine

__global__ __launch_bounds__(256) void ffn_kernel(
    const float* __restrict__ x,
    const unsigned short* __restrict__ pg,
    const unsigned short* __restrict__ pl,
    const int* __restrict__ counts,
    const int* __restrict__ btok,
    const float* __restrict__ bwt,
    const float* __restrict__ pes,
    float* __restrict__ out)
{
    __shared__ alignas(32) unsigned short A1[(DQ / 32) * 32 * 16];  // 8 KB
    __shared__ alignas(32) unsigned short A2[(HQ / 32) * 32 * 16];  // 16 KB
    __shared__ int   tok_s[16];
    __shared__ float wt_s[16];

    int e    = blockIdx.x / MAX_TILES;
    int tile = blockIdx.x % MAX_TILES;
    int cnt  = counts[e];
    if (tile * 16 >= cnt) return;

    int tid = threadIdx.x;
    if (tid < 16) {
        int idx = tile * 16 + tid;
        int ok = idx < cnt;
        tok_s[tid] = btok[e * CAPQ + (ok ? idx : 0)];
        wt_s[tid]  = ok ? bwt[e * CAPQ + idx] : 0.0f;
    }
    __syncthreads();

    // ---- pack A1: gather 16 token rows of x, convert to bf16 in A-operand layout ----
    {
        int kc = tid >> 5, L = tid & 31;
        int m = L & 15;
        const float* xr = x + (long)tok_s[m] * DQ;
        unsigned short* dst = &A1[(kc * 32 + L) * 16];
#pragma unroll
        for (int r = 0; r < 16; ++r)
            dst[r] = f2bf(xr[kc * 32 + a_koff(L, r)]);
    }
    __syncthreads();

    int w = tid >> 5;      // wave id (uniform within wave)
    int lane = tid & 31;

    const v16bf* pgv = reinterpret_cast<const v16bf*>(pg);
    const v16bf* plv = reinterpret_cast<const v16bf*>(pl);

    // ---- GEMM1 + gated activation + scatter to A2 ----
    for (int fi = 0; fi < 4; ++fi) {
        int ft = w * 4 + fi;                     // 32 f-tiles over 8 waves
        v8f acc1 = {0.f, 0.f, 0.f, 0.f, 0.f, 0.f, 0.f, 0.f};
        v8f acc2 = {0.f, 0.f, 0.f, 0.f, 0.f, 0.f, 0.f, 0.f};
        long base1 = ((((long)e * 2 + 0) * 32 + ft) * 8) * 32;
        long base2 = ((((long)e * 2 + 1) * 32 + ft) * 8) * 32;
#pragma unroll
        for (int kc = 0; kc < 8; ++kc) {
            v16bf a  = *reinterpret_cast<const v16bf*>(&A1[(kc * 32 + lane) * 16]);
            v16bf b1 = pgv[base1 + (long)kc * 32 + lane];
            v16bf b2 = pgv[base2 + (long)kc * 32 + lane];
            acc1 = __builtin_amdgcn_wmma_f32_16x16x32_bf16(
                       false, a, false, b1, (short)0, acc1, false, false);
            acc2 = __builtin_amdgcn_wmma_f32_16x16x32_bf16(
                       false, a, false, b2, (short)0, acc2, false, false);
        }
        // act = gelu(x1) * x2 ; scatter into A2 (A-operand layout, K = f)
        int f   = ft * 16 + (lane & 15);         // constant per lane
        int kc2 = f >> 5;
        int kk  = f & 31;
        int rp  = (kk & 7) + ((kk >> 4) << 3);
        int Lhi = ((kk >> 3) & 1) << 4;
#pragma unroll
        for (int i = 0; i < 8; ++i) {
            int m = i + ((lane >> 4) << 3);      // C/D layout row
            float g = gelu_tanh(acc1[i]) * acc2[i];
            A2[(kc2 * 32 + (m + Lhi)) * 16 + rp] = f2bf(g);
        }
    }
    __syncthreads();

    // ---- GEMM2: act[16x512] x Wl[512x256], scaled atomic combine ----
    float se = pes[e];
    for (int di = 0; di < 2; ++di) {
        int dt = w * 2 + di;                     // 16 d-tiles over 8 waves
        v8f acc = {0.f, 0.f, 0.f, 0.f, 0.f, 0.f, 0.f, 0.f};
        long base = (((long)e * 16 + dt) * 16) * 32;
        __builtin_prefetch(&plv[base + lane], 0, 1);
#pragma unroll
        for (int kc2 = 0; kc2 < 16; ++kc2) {
            v16bf a = *reinterpret_cast<const v16bf*>(&A2[(kc2 * 32 + lane) * 16]);
            v16bf b = plv[base + (long)kc2 * 32 + lane];
            acc = __builtin_amdgcn_wmma_f32_16x16x32_bf16(
                      false, a, false, b, (short)0, acc, false, false);
        }
        int d = dt * 16 + (lane & 15);
#pragma unroll
        for (int i = 0; i < 8; ++i) {
            int m = i + ((lane >> 4) << 3);
            float val = se * wt_s[m] * acc[i];
            atomicAdd(&out[(long)tok_s[m] * DQ + d], val);
        }
    }
}

// ---------- launch ----------

extern "C" void kernel_launch(void* const* d_in, const int* in_sizes, int n_in,
                              void* d_out, int out_size, void* d_ws, size_t ws_size,
                              hipStream_t stream) {
    const float* x   = (const float*)d_in[0];  // (G,S,D)
    const float* wr  = (const float*)d_in[1];  // (D,E)
    const float* wg  = (const float*)d_in[2];  // (E,2,H,D)
    const float* wl  = (const float*)d_in[3];  // (E,H,D)
    const float* pes = (const float*)d_in[4];  // (E,)
    const float* rsc = (const float*)d_in[5];  // (D,)
    // d_in[6] = k (always 2)
    (void)in_sizes; (void)n_in; (void)out_size; (void)ws_size;

    char* ws = (char*)d_ws;
    int*            counts = (int*)ws;                                   // 8 ints (256B slot)
    int*            btok   = (int*)(ws + 256);                           // E*CAPQ ints   = 1 MB
    float*          bwt    = (float*)(ws + 256 + (size_t)EQ * CAPQ * 4); // E*CAPQ floats = 1 MB
    unsigned short* pg     = (unsigned short*)(ws + 256 + 2ull * EQ * CAPQ * 4);  // 4 MB
    unsigned short* pl     = pg + (size_t)EQ * 2 * HQ * DQ;                        // 2 MB
    float* out = (float*)d_out;

    zero_kernel <<<256,  256, 0, stream>>>(out, BQ * DQ, counts);
    pack_gating <<<2048, 256, 0, stream>>>(wg, pg);
    pack_linear <<<1024, 256, 0, stream>>>(wl, pl);
    router_kernel<<<BQ,  256, 0, stream>>>(x, wr, rsc, counts, btok, bwt);
    ffn_kernel  <<<EQ * MAX_TILES, 256, 0, stream>>>(x, pg, pl, counts, btok, bwt, pes, out);
}
